// AdaAPPNP_86964497809759
// MI455X (gfx1250) — matrix operand order repaired and checked by reading, not compiled, not run on hardware
//
#include <hip/hip_runtime.h>
#include <hip/hip_bf16.h>
#include <stdint.h>

// ---------------------------------------------------------------------------
// Gated APPNP forward for MI455X (gfx1250, wave32).
//  - Dense GEMMs via v_wmma_f32_16x16x32_bf16, f32 accumulate. Weights are
//    pre-transposed/converted to bf16 [N][K]; B tiles staged to LDS with
//    GLOBAL_LOAD_ASYNC_TO_LDS_B128 (ASYNCcnt) when available. A tiles staged
//    as float2 -> v_cvt_pk_bf16_f32 -> b32 LDS stores. Fragments read from
//    LDS as 16B chunks (ds_load_b128), 2 WMMAs per barrier pair.
//  - SpMM propagation as edge-parallel gather/scatter with f32 global atomics
//    (node matrices are 25.6MB => L2-resident on the 192MB L2).
// ---------------------------------------------------------------------------

typedef __attribute__((ext_vector_type(16))) __bf16 v16bf;
typedef __attribute__((ext_vector_type(8)))  __bf16 v8bf;
typedef __attribute__((ext_vector_type(2)))  __bf16 v2bf;
typedef __attribute__((ext_vector_type(8)))  float  v8f;
typedef int v4i __attribute__((vector_size(16)));   // 128-bit chunk type

#if defined(__gfx1250__) && __has_builtin(__builtin_amdgcn_global_load_async_to_lds_b128)
#define HAS_ASYNC_LDS 1
#else
#define HAS_ASYNC_LDS 0
#endif

static __device__ __forceinline__ void copy16_g2l(const __bf16* gp, __bf16* lp) {
#if HAS_ASYNC_LDS
  __builtin_amdgcn_global_load_async_to_lds_b128((v4i*)gp, (v4i*)lp, 0, 0);
#else
  *(v8bf*)lp = *(const v8bf*)gp;
#endif
}

static __device__ __forceinline__ void wait_async_copies() {
#if HAS_ASYNC_LDS
#if __has_builtin(__builtin_amdgcn_s_wait_asynccnt)
  __builtin_amdgcn_s_wait_asynccnt(0);
#else
  asm volatile("s_wait_asynccnt 0x0" ::: "memory");
#endif
#endif
}

static __device__ __forceinline__ v16bf ld_frag(const __bf16* p0, const __bf16* p1) {
  v8bf lo = *(const v8bf*)p0;   // 16 contiguous bytes
  v8bf hi = *(const v8bf*)p1;   // 16 contiguous bytes
  return __builtin_shufflevector(lo, hi, 0, 1, 2, 3, 4, 5, 6, 7,
                                 8, 9, 10, 11, 12, 13, 14, 15);
}

// ---------------------- utility kernels ------------------------------------

__global__ void zero_f32_kernel(float* __restrict__ p, long long n) {
  long long i = (long long)blockIdx.x * blockDim.x + threadIdx.x;
  if (i < n) p[i] = 0.0f;
}

__global__ void degree_kernel(const int* __restrict__ dst, float* __restrict__ deg, int E) {
  int e = blockIdx.x * blockDim.x + threadIdx.x;
  if (e < E) atomicAdd(&deg[dst[e]], 1.0f);
}

__global__ void norm_kernel(float* __restrict__ deg, int n) {
  int i = blockIdx.x * blockDim.x + threadIdx.x;
  if (i < n) deg[i] = rsqrtf(fmaxf(deg[i], 1.0f));
}

// W[K,N] f32 -> Wt[N,K] bf16 (small matrices; coalesced on the output).
__global__ void transpose_cvt_kernel(const float* __restrict__ Wsrc,
                                     __bf16* __restrict__ Wt, int K, int N) {
  int o = blockIdx.x * blockDim.x + threadIdx.x;
  if (o >= K * N) return;
  int n = o / K;
  int k = o - n * K;
  Wt[o] = (__bf16)Wsrc[(size_t)k * N + n];
}

// ---------------------- WMMA GEMM ------------------------------------------
// C[M,N] = op(A[M,K] @ B[K,N] + bias), B given transposed bf16 [N][K].
// Grid: (M/16). Block: (32, NW) with NW = N/16. Requires M%16==0, K%64==0.
#define KT 64
#define LDS_STRIDE 72   // 72 bf16 = 144 B row stride (multiple of 16 B)

template <int NW>
__global__ __launch_bounds__(32 * NW)
void gemm_bf16_wmma(const float* __restrict__ A,
                    const __bf16* __restrict__ Bt,   // [N][K] bf16
                    const float* __restrict__ bias,
                    float* __restrict__ C,
                    int M, int N, int K, int do_relu) {
  __shared__ __attribute__((aligned(16))) __bf16 As[16][LDS_STRIDE];      // 16 rows x KT
  __shared__ __attribute__((aligned(16))) __bf16 Bs[NW][16][LDS_STRIDE];  // [wave][col][k]

  const int m0   = blockIdx.x * 16;
  const int wy   = threadIdx.y;                 // column-tile wave
  const int n0   = wy * 16;
  const int lane = threadIdx.x;                 // 0..31
  const int r    = lane & 15;
  const int hi   = lane >> 4;
  const int tid  = wy * 32 + lane;

  v8f acc = {};

  for (int kk = 0; kk < K; kk += KT) {
    // Stage A tile (16 x KT f32 -> bf16): float2 pairs -> packed b32 stores.
#pragma unroll
    for (int i = 0; i < 512 / (32 * NW); ++i) {
      int idx = tid + i * 32 * NW;       // 0..511 pair index
      int row = idx >> 5;
      int c2  = idx & 31;                // pair within row
      float2 f = *(const float2*)(A + (size_t)(m0 + row) * K + kk + 2 * c2);
      v2bf p;
      p[0] = (__bf16)f.x;
      p[1] = (__bf16)f.y;
      *(v2bf*)&As[row][2 * c2] = p;
    }
    // Stage this wave's B tile: 16 cols x KT bf16 = 128 chunks of 16B,
    // contiguous in both global (Bt row-major [N][K]) and LDS.
#pragma unroll
    for (int i = 0; i < 4; ++i) {
      int idx = i * 32 + lane;           // 0..127 chunk index
      int col = idx >> 3;
      int seg = idx & 7;
      copy16_g2l(Bt + (size_t)(n0 + col) * K + kk + seg * 8,
                 &Bs[wy][col][seg * 8]);
    }
    wait_async_copies();
    __syncthreads();

    // Two 16x16x32 WMMAs per staged chunk. Fragment element mapping
    // (CDNA5 ISA): A lane r/hi, elem e -> k = ko + hi*8 + e (+16 for e>=8);
    // B (per-col contiguous in LDS) elem e -> k = ko + hi*8 + e (+16 for e>=8), col = r.
#pragma unroll
    for (int ko = 0; ko < KT; ko += 32) {
      v16bf a = ld_frag(&As[r][ko + hi * 8], &As[r][ko + hi * 8 + 16]);
      v16bf b = ld_frag(&Bs[wy][r][ko + hi * 8], &Bs[wy][r][ko + hi * 8 + 16]);
      acc = __builtin_amdgcn_wmma_f32_16x16x32_bf16(
          false, a, false, b, (short)0, acc, false, false);
    }
    __syncthreads();
  }

  // Epilogue: single hoisted bias load, then 8 coalesced stores.
  const int col = n0 + r;
  float bv = 0.0f;
  if (bias) bv = bias[col];
#pragma unroll
  for (int v = 0; v < 8; ++v) {
    int row = m0 + v + hi * 8;
    float x = acc[v] + bv;
    if (do_relu) x = fmaxf(x, 0.0f);
    C[(size_t)row * N + col] = x;
  }
}

// ---------------------- row softmax over 64 classes ------------------------
__global__ void softmax64_kernel(const float* __restrict__ in,
                                 float* __restrict__ out, int n) {
  int node = blockIdx.x * (blockDim.x >> 5) + (threadIdx.x >> 5);
  int lane = threadIdx.x & 31;
  if (node >= n) return;
  const float* row = in + (size_t)node * 64;
  float a = row[lane];
  float b = row[lane + 32];
  float m = fmaxf(a, b);
#pragma unroll
  for (int off = 16; off > 0; off >>= 1) m = fmaxf(m, __shfl_xor(m, off, 32));
  float ea = __expf(a - m);
  float eb = __expf(b - m);
  float s = ea + eb;
#pragma unroll
  for (int off = 16; off > 0; off >>= 1) s += __shfl_xor(s, off, 32);
  float inv = 1.0f / s;
  float* orow = out + (size_t)node * 64;
  orow[lane]      = ea * inv;
  orow[lane + 32] = eb * inv;
}

// ---------------------- fused dual SpMM ------------------------------------
__global__ void spmm_dual_kernel(const int* __restrict__ src,
                                 const int* __restrict__ dstv,
                                 const float* __restrict__ nrm,
                                 const float* __restrict__ lg,
                                 const float* __restrict__ hc,
                                 float* __restrict__ agg,
                                 float* __restrict__ hacc, int E) {
  long long gid = (long long)blockIdx.x * blockDim.x + threadIdx.x;
  long long e   = gid >> 5;
  int       l   = (int)(gid & 31);
  if (e >= E) return;
  int s = src[e];
  int d = dstv[e];
  float w = nrm[s] * nrm[d];
  size_t sb = (size_t)s * 64 + l * 2;
  size_t db = (size_t)d * 64 + l * 2;
  float2 l2 = *(const float2*)(lg + sb);
  float2 h2 = *(const float2*)(hc + sb);
  atomicAdd(agg  + db,     l2.x * w);
  atomicAdd(agg  + db + 1, l2.y * w);
  atomicAdd(hacc + db,     h2.x * w);
  atomicAdd(hacc + db + 1, h2.y * w);
}

// ---------------------- gate + blend ---------------------------------------
__global__ void combine_kernel(const float* __restrict__ lg,
                               const float* __restrict__ agg,
                               const float* __restrict__ sh,
                               const float* __restrict__ h0,
                               float* __restrict__ hout, int n) {
  int node = blockIdx.x * (blockDim.x >> 5) + (threadIdx.x >> 5);
  int lane = threadIdx.x & 31;
  if (node >= n) return;
  size_t base = (size_t)node * 64 + lane * 2;
  float2 l2 = *(const float2*)(lg  + base);
  float2 a2 = *(const float2*)(agg + base);
  float dot = l2.x * a2.x + l2.y * a2.y;
#pragma unroll
  for (int off = 16; off > 0; off >>= 1) dot += __shfl_xor(dot, off, 32);
  float z = 1.0f / (1.0f + __expf(-dot));
  float2 s2 = *(const float2*)(sh + base);
  float2 h2 = *(const float2*)(h0 + base);
  float2 o;
  o.x = z * s2.x + (1.0f - z) * h2.x;
  o.y = z * s2.y + (1.0f - z) * h2.y;
  *(float2*)(hout + base) = o;
}

// ---------------------------------------------------------------------------

static inline void launch_zero(float* p, long long n, hipStream_t s) {
  int blocks = (int)((n + 255) / 256);
  zero_f32_kernel<<<blocks, 256, 0, s>>>(p, n);
}

extern "C" void kernel_launch(void* const* d_in, const int* in_sizes, int n_in,
                              void* d_out, int out_size, void* d_ws, size_t ws_size,
                              hipStream_t stream) {
  const float* features = (const float*)d_in[0];
  const float* W1       = (const float*)d_in[1];
  const float* b1       = (const float*)d_in[2];
  const float* W2       = (const float*)d_in[3];
  const float* b2       = (const float*)d_in[4];
  const float* Wy       = (const float*)d_in[5];
  const int*   ei       = (const int*)d_in[6];

  const int H   = in_sizes[2];              // 128
  const int C   = in_sizes[4];              // 64
  const int INF = in_sizes[1] / H;          // 512
  const int NS  = in_sizes[0] / INF;        // 100000
  const int E   = in_sizes[6] / 2;          // 3.2M
  const int* srcv = ei;
  const int* dstv = ei + E;

  // Workspace layout (f32 region, then bf16 transposed weights).
  float* W    = (float*)d_ws;
  float* nrm  = W;                                  // NS (padded to 102400)
  float* h0   = W + 102400;                         // NS*C
  float* lgA  = h0  + (size_t)NS * C;               // NS*C
  float* lgB  = lgA + (size_t)NS * C;               // NS*C
  float* hA   = lgB + (size_t)NS * C;               // NS*C
  float* hbig = hA + (size_t)NS * C;                // NS*H (h; reused below)
  float* hB   = hbig;                               // NS*C (alias, post-h0)
  float* tmp  = hbig + (size_t)NS * C;              // NS*C (alias, post-h0)
  __bf16* wt1 = (__bf16*)(hbig + (size_t)NS * H);   // [H][INF]
  __bf16* wty = wt1 + (size_t)H * INF;              // [C][INF]
  __bf16* wt2 = wty + (size_t)C * INF;              // [C][H]
  (void)ws_size; (void)n_in;

  // 0) Weights -> transposed bf16.
  transpose_cvt_kernel<<<(INF * H + 255) / 256, 256, 0, stream>>>(W1, wt1, INF, H);
  transpose_cvt_kernel<<<(INF * C + 255) / 256, 256, 0, stream>>>(Wy, wty, INF, C);
  transpose_cvt_kernel<<<(H * C + 255) / 256, 256, 0, stream>>>(W2, wt2, H, C);

  // 1) Degrees -> symmetric-norm factors.
  launch_zero(nrm, NS, stream);
  degree_kernel<<<(E + 255) / 256, 256, 0, stream>>>(dstv, nrm, E);
  norm_kernel<<<(NS + 255) / 256, 256, 0, stream>>>(nrm, NS);

  // 2) h = relu(features @ W1 + b1)          [NS,H]
  gemm_bf16_wmma<8><<<NS / 16, dim3(32, 8), 0, stream>>>(
      features, wt1, b1, hbig, NS, H, INF, 1);

  // 3) logits = softmax(features @ Wy)       [NS,C]  (scores staged in lgB)
  gemm_bf16_wmma<4><<<NS / 16, dim3(32, 4), 0, stream>>>(
      features, wty, nullptr, lgB, NS, C, INF, 0);
  softmax64_kernel<<<(NS + 7) / 8, 256, 0, stream>>>(lgB, lgA, NS);

  // 4) h0 = h @ W2 + b2                      [NS,C]
  gemm_bf16_wmma<4><<<NS / 16, dim3(32, 4), 0, stream>>>(
      hbig, wt2, b2, h0, NS, C, H, 0);

  // 5) K=10 gated APPNP hops. hbig region is now free -> hB/tmp aliases.
  const float* hc    = h0;
  float*       lgcur = lgA;
  float*       lgnxt = lgB;
  const long long nc = (long long)NS * C;
  const long long ethreads = (long long)E * 32;
  const int eblocks = (int)((ethreads + 255) / 256);
  for (int t = 0; t < 10; ++t) {
    float* hdst = (t == 9) ? (float*)d_out : ((t & 1) ? hB : hA);
    launch_zero(lgnxt, nc, stream);
    launch_zero(tmp,   nc, stream);
    spmm_dual_kernel<<<eblocks, 256, 0, stream>>>(srcv, dstv, nrm, lgcur, hc,
                                                  lgnxt, tmp, E);
    combine_kernel<<<(NS + 7) / 8, 256, 0, stream>>>(lgcur, lgnxt, tmp, h0,
                                                     hdst, NS);
    hc = hdst;
    float* sw = lgcur; lgcur = lgnxt; lgnxt = sw;
  }
  (void)out_size;
}
